// MultiheadAttention_46986942218984
// MI455X (gfx1250) — matrix operand order repaired and checked by reading, not compiled, not run on hardware
//
#include <hip/hip_runtime.h>
#include <hip/hip_bf16.h>
#include <math.h>

// ---------------------------------------------------------------------------
// MI455X (gfx1250) fused pruned-attention pipeline.
// All matmuls: C[M,N] = A[M,K] @ Bt[N,K]^T via v_wmma_f32_16x16x32_bf16
// (f32 accumulate, bf16 operands). The two SxSxS matmuls (70% of FLOPs) use
// double-buffered async global->LDS staging (ASYNCcnt) when the toolchain
// exposes the gfx1250 async builtins; otherwise a sync LDS-staged fallback.
// ---------------------------------------------------------------------------

typedef __bf16 bf16_t;
typedef __attribute__((ext_vector_type(16))) __bf16 v16bf;
typedef __attribute__((ext_vector_type(8)))  __bf16 v8bf;
typedef __attribute__((ext_vector_type(8)))  float  v8f;
typedef __attribute__((ext_vector_type(4)))  int    v4i;

static constexpr int Bn = 4, Sn = 512, En = 512, Hn = 8, HD = 64;
static constexpr int F3 = 3 * En;            // 1536
static constexpr int BH = Bn * Hn;           // 32

// ---- gfx1250 async global->LDS support (probe via __has_builtin) ----------
#if defined(__has_builtin)
#if __has_builtin(__builtin_amdgcn_global_load_async_to_lds_b128) && \
    __has_builtin(__builtin_amdgcn_s_wait_asynccnt)
#define HAVE_ASYNC_LDS 1
#endif
#endif
#ifndef HAVE_ASYNC_LDS
#define HAVE_ASYNC_LDS 0
#endif

#if HAVE_ASYNC_LDS
// Builtin signature (clang): (v4i addrspace(1)*, v4i addrspace(3)*, Ii, Ii)
typedef __attribute__((address_space(1))) v4i as1_v4i;
typedef __attribute__((address_space(3))) v4i as3_v4i;
__device__ inline void async_ld16(const void* g, void* l) {
  // GLOBAL_LOAD_ASYNC_TO_LDS_B128: per-lane 16B, tracked by ASYNCcnt
  __builtin_amdgcn_global_load_async_to_lds_b128(
      (as1_v4i*)(size_t)g, (as3_v4i*)l, 0, 0);
}
#endif

// ---- WMMA fragment loaders (wave32 layouts per CDNA5 ISA 7.12.2) ----------
__device__ inline v16bf load_a_frag(const bf16_t* __restrict__ A, int lda,
                                    int row0, int k0, int lane) {
  const int m    = lane & 15;
  const int half = lane >> 4;
  const bf16_t* p = A + (size_t)(row0 + m) * lda + k0 + half * 8;
  union { v16bf v; v8bf h[2]; } u;
  u.h[0] = *(const v8bf*)(p);
  u.h[1] = *(const v8bf*)(p + 16);
  return u.v;
}
__device__ inline v16bf load_b_frag(const bf16_t* __restrict__ Bt, int ldb,
                                    int col0, int k0, int lane) {
  const int n    = lane & 15;
  const int half = lane >> 4;
  const bf16_t* p = Bt + (size_t)(col0 + n) * ldb + k0 + half * 8;
  union { v16bf v; v8bf h[2]; } u;
  u.h[0] = *(const v8bf*)(p);
  u.h[1] = *(const v8bf*)(p + 16);
  return u.v;
}

// ---- Direct-load batched WMMA GEMM (qkv, K=64 projections, output) --------
template <typename CT, bool NANZ>
__global__ void __launch_bounds__(256)
wmma_gemm(const bf16_t* __restrict__ Abase, const bf16_t* __restrict__ Btbase,
          CT* __restrict__ Cbase, int M, int N, int K,
          long strideA, long strideB, long strideC, float scale) {
  const int lane   = threadIdx.x & 31;
  const int wave   = threadIdx.x >> 5;
  const int tilesN = N >> 6;
  const int waveId = blockIdx.x * 8 + wave;
  const int tm     = waveId / tilesN;
  const int tn     = waveId - tm * tilesN;
  if (tm * 64 >= M) return;                      // wave-uniform

  const int batch = blockIdx.y;
  const bf16_t* A  = Abase  + (size_t)batch * strideA;
  const bf16_t* Bt = Btbase + (size_t)batch * strideB;
  CT*           C  = Cbase  + (size_t)batch * strideC;
  const int row0 = tm * 64, col0 = tn * 64;

  v8f acc[4][4] = {};
  for (int k0 = 0; k0 < K; k0 += 32) {
    if (k0 + 32 < K) {
      __builtin_prefetch(A  + (size_t)(row0 + (lane & 15)) * K + k0 + 32, 0, 1);
      __builtin_prefetch(Bt + (size_t)(col0 + (lane & 15)) * K + k0 + 32, 0, 1);
    }
    v16bf a[4], b[4];
#pragma unroll
    for (int i = 0; i < 4; ++i) a[i] = load_a_frag(A, K, row0 + 16 * i, k0, lane);
#pragma unroll
    for (int j = 0; j < 4; ++j) b[j] = load_b_frag(Bt, K, col0 + 16 * j, k0, lane);
#pragma unroll
    for (int i = 0; i < 4; ++i)
#pragma unroll
      for (int j = 0; j < 4; ++j)
        acc[i][j] = __builtin_amdgcn_wmma_f32_16x16x32_bf16(
            false, a[i], false, b[j], (short)0, acc[i][j], false, false);
  }
  const int nlane = lane & 15;
  const int mo    = (lane >> 4) * 8;
#pragma unroll
  for (int i = 0; i < 4; ++i)
#pragma unroll
    for (int j = 0; j < 4; ++j)
#pragma unroll
      for (int r = 0; r < 8; ++r) {
        float v = acc[i][j][r] * scale;
        if (NANZ && __builtin_isnan(v)) v = 0.0f;
        C[(size_t)(row0 + 16 * i + mo + r) * N + (col0 + 16 * j + nlane)] = (CT)v;
      }
}

// ---- LDS-staged WMMA GEMM, fixed 512x512x512, bf16 out --------------------
// Block tile 128x256, 8 waves (2x4), wave tile 64x64. LDS double-buffered,
// filled by async global->LDS b128 copies, drained by ds_load_b128 frags.
template <bool NANZ>
__global__ void __launch_bounds__(256)
wmma_gemm_lds(const bf16_t* __restrict__ Abase, const bf16_t* __restrict__ Btbase,
              bf16_t* __restrict__ Cbase, float scale) {
  constexpr int Kc = 512, Nc = 512;
  constexpr int LDA = 40;                        // padded LDS row stride (elems)
  constexpr int NBUF = HAVE_ASYNC_LDS ? 2 : 1;
  __shared__ __align__(16) bf16_t shA[NBUF][128 * LDA];  // 10 KB per buf
  __shared__ __align__(16) bf16_t shB[NBUF][256 * LDA];  // 20 KB per buf

  const int tid  = threadIdx.x;
  const int lane = tid & 31;
  const int wave = tid >> 5;
  const int bm = blockIdx.x >> 1, bn = blockIdx.x & 1;
  const int row0 = bm * 128, col0 = bn * 256;
  const int batch = blockIdx.y;
  const bf16_t* A  = Abase  + (size_t)batch * Kc * Nc;
  const bf16_t* Bt = Btbase + (size_t)batch * Kc * Nc;
  bf16_t*       C  = Cbase  + (size_t)batch * Kc * Nc;

  const int wm = wave >> 2, wn = wave & 3;       // 2 x 4 wave grid
  const int wrow = wm * 64, wcol = wn * 64;

  auto issueA = [&](bf16_t* dst, int k0) {       // 128 rows x 32 K = 512 x 16B
#pragma unroll
    for (int it = 0; it < 2; ++it) {
      int c = tid + it * 256;
      int r = c >> 2, p = c & 3;
      const bf16_t* g = A + (size_t)(row0 + r) * Kc + k0 + p * 8;
      bf16_t* l = dst + r * LDA + p * 8;
#if HAVE_ASYNC_LDS
      async_ld16(g, l);
#else
      *(v8bf*)l = *(const v8bf*)g;
#endif
    }
  };
  auto issueB = [&](bf16_t* dst, int k0) {       // 256 rows x 32 K = 1024 x 16B
#pragma unroll
    for (int it = 0; it < 4; ++it) {
      int c = tid + it * 256;
      int r = c >> 2, p = c & 3;
      const bf16_t* g = Bt + (size_t)(col0 + r) * Kc + k0 + p * 8;
      bf16_t* l = dst + r * LDA + p * 8;
#if HAVE_ASYNC_LDS
      async_ld16(g, l);
#else
      *(v8bf*)l = *(const v8bf*)g;
#endif
    }
  };

  v8f acc[4][4] = {};
#if HAVE_ASYNC_LDS
  issueA(shA[0], 0);
  issueB(shB[0], 0);
#endif
  for (int ks = 0; ks < Kc / 32; ++ks) {
#if HAVE_ASYNC_LDS
    const int cur = ks & 1;
    __builtin_amdgcn_s_wait_asynccnt(0);         // own async copies landed
    __syncthreads();                             // whole tile visible; step k-1 reads done
    if (ks + 1 < Kc / 32) {                      // prefetch next K-slab into other buffer
      issueA(shA[cur ^ 1], (ks + 1) * 32);
      issueB(shB[cur ^ 1], (ks + 1) * 32);
    }
#else
    const int cur = 0;
    __syncthreads();                             // protect previous reads
    issueA(shA[0], ks * 32);
    issueB(shB[0], ks * 32);
    __syncthreads();
#endif
    v16bf a[4], b[4];
    const int ml = lane & 15;
    const int half = lane >> 4;
#pragma unroll
    for (int i = 0; i < 4; ++i) {                // ds_load_b128 x2 per frag
      const bf16_t* p = &shA[cur][(wrow + 16 * i + ml) * LDA + half * 8];
      union { v16bf v; v8bf h[2]; } u;
      u.h[0] = *(const v8bf*)(p);
      u.h[1] = *(const v8bf*)(p + 16);
      a[i] = u.v;
    }
#pragma unroll
    for (int j = 0; j < 4; ++j) {
      const bf16_t* p = &shB[cur][(wcol + 16 * j + ml) * LDA + half * 8];
      union { v16bf v; v8bf h[2]; } u;
      u.h[0] = *(const v8bf*)(p);
      u.h[1] = *(const v8bf*)(p + 16);
      b[j] = u.v;
    }
#pragma unroll
    for (int i = 0; i < 4; ++i)
#pragma unroll
      for (int j = 0; j < 4; ++j)
        acc[i][j] = __builtin_amdgcn_wmma_f32_16x16x32_bf16(
            false, a[i], false, b[j], (short)0, acc[i][j], false, false);
  }

  const int nlane = lane & 15;
  const int mo    = (lane >> 4) * 8;
  const int crow0 = row0 + wrow, ccol0 = col0 + wcol;
#pragma unroll
  for (int i = 0; i < 4; ++i)
#pragma unroll
    for (int j = 0; j < 4; ++j)
#pragma unroll
      for (int r = 0; r < 8; ++r) {
        float v = acc[i][j][r] * scale;
        if (NANZ && __builtin_isnan(v)) v = 0.0f;
        C[(size_t)(crow0 + 16 * i + mo + r) * Nc + (ccol0 + 16 * j + nlane)] = (bf16_t)v;
      }
}

// ---- f32 -> bf16 conversion ----------------------------------------------
__global__ void __launch_bounds__(256)
f32_to_bf16(const float* __restrict__ in, bf16_t* __restrict__ out, int n) {
  int i = blockIdx.x * blockDim.x + threadIdx.x;
  if (i < n) out[i] = (bf16_t)in[i];
}

// ---- prune q/k (2x2 tile mask vs mean norm) + v (top-S/2 rows) ------------
__global__ void __launch_bounds__(256)
prune_qkv(const float* __restrict__ qkv, const float* __restrict__ bqkv,
          const float* __restrict__ bias_t,
          bf16_t* __restrict__ qp, bf16_t* __restrict__ kp,
          bf16_t* __restrict__ vp) {
  const int bh = blockIdx.x;
  const int b = bh >> 3, h = bh & 7;
  const int tid = threadIdx.x;

  __shared__ float tileN[8192];
  __shared__ float part[256];
  __shared__ float meanSh, threshSh;
  __shared__ float rowN[512];

  const float* bias = bias_t + (size_t)bh * Sn * HD;
  const size_t rowBase = (size_t)b * Sn;

  for (int pass = 0; pass < 2; ++pass) {
    const int off = h * 3 * HD + pass * HD;
    float partial = 0.f;
    for (int t = tid; t < 8192; t += 256) {
      int ts = t >> 5, td = t & 31;
      float s2 = 0.f;
#pragma unroll
      for (int dy = 0; dy < 2; ++dy)
#pragma unroll
        for (int dx = 0; dx < 2; ++dx) {
          int s = ts * 2 + dy, d = td * 2 + dx;
          float wv = qkv[(rowBase + s) * F3 + off + d] + bqkv[off + d];
          float u  = wv + bias[s * HD + d];
          s2 += u * u;
        }
      float nr = sqrtf(s2);
      tileN[t] = nr;
      partial += nr;
    }
    part[tid] = partial;
    __syncthreads();
    if (tid == 0) {
      float tot = 0.f;
      for (int i = 0; i < 256; ++i) tot += part[i];
      meanSh = tot / 8192.0f;
    }
    __syncthreads();
    bf16_t* out = (pass ? kp : qp) + (size_t)bh * Sn * HD;
    const float mean = meanSh;
    for (int e = tid; e < Sn * HD; e += 256) {
      int s = e >> 6, d = e & 63;
      float nr = tileN[(s >> 1) * 32 + (d >> 1)];
      float wv = qkv[(rowBase + s) * F3 + off + d] + bqkv[off + d];
      out[e] = (bf16_t)((nr >= mean) ? wv : 0.0f);
    }
    __syncthreads();
  }

  const int offv = h * 3 * HD + 2 * HD;
  for (int s = tid; s < Sn; s += 256) {
    float s2 = 0.f;
    for (int d = 0; d < HD; ++d) {
      float wv = qkv[(rowBase + s) * F3 + offv + d] + bqkv[offv + d];
      s2 += wv * wv;
    }
    rowN[s] = sqrtf(s2);
  }
  __syncthreads();
  for (int s = tid; s < Sn; s += 256) {
    float mine = rowN[s];
    int rank = 0;
    for (int j = 0; j < Sn; ++j) {
      float o = rowN[j];
      rank += (o < mine) || (o == mine && j < s);
    }
    if (rank == Sn / 2) threshSh = mine;
  }
  __syncthreads();
  bf16_t* vout = vp + (size_t)bh * Sn * HD;
  const float th = threshSh;
  for (int e = tid; e < Sn * HD; e += 256) {
    int s = e >> 6, d = e & 63;
    float wv = qkv[(rowBase + s) * F3 + offv + d] + bqkv[offv + d];
    vout[e] = (bf16_t)((rowN[s] >= th) ? wv : 0.0f);
  }
}

// ---- prune WO and write transposed [hd, S] bf16 ---------------------------
__global__ void __launch_bounds__(256)
prune_wo(const float* __restrict__ WO, const float* __restrict__ bias_t,
         bf16_t* __restrict__ wopT) {
  const int bh = blockIdx.x;
  const int tid = threadIdx.x;
  __shared__ float tileN[8192];
  __shared__ float part[256];
  __shared__ float meanSh;

  const float* w    = WO     + (size_t)bh * Sn * HD;
  const float* bias = bias_t + (size_t)bh * Sn * HD;
  float partial = 0.f;
  for (int t = tid; t < 8192; t += 256) {
    int ts = t >> 5, td = t & 31;
    float s2 = 0.f;
#pragma unroll
    for (int dy = 0; dy < 2; ++dy)
#pragma unroll
      for (int dx = 0; dx < 2; ++dx) {
        int s = ts * 2 + dy, d = td * 2 + dx;
        float u = w[s * HD + d] + bias[s * HD + d];
        s2 += u * u;
      }
    float nr = sqrtf(s2);
    tileN[t] = nr;
    partial += nr;
  }
  part[tid] = partial;
  __syncthreads();
  if (tid == 0) {
    float tot = 0.f;
    for (int i = 0; i < 256; ++i) tot += part[i];
    meanSh = tot / 8192.0f;
  }
  __syncthreads();
  bf16_t* out = wopT + (size_t)bh * HD * Sn;
  const float mean = meanSh;
  for (int e = tid; e < Sn * HD; e += 256) {
    int s = e >> 6, d = e & 63;
    float nr = tileN[(s >> 1) * 32 + (d >> 1)];
    out[(size_t)d * Sn + s] = (bf16_t)((nr >= mean) ? w[e] : 0.0f);
  }
}

// ---------------------------------------------------------------------------
extern "C" void kernel_launch(void* const* d_in, const int* in_sizes, int n_in,
                              void* d_out, int out_size, void* d_ws, size_t ws_size,
                              hipStream_t stream) {
  const float* x      = (const float*)d_in[0];   // [B,S,E]
  const float* Wqkv   = (const float*)d_in[1];   // [3E,E]
  const float* bqkv   = (const float*)d_in[2];   // [3E]
  const float* WO     = (const float*)d_in[3];   // [B,H,S,hd]
  const float* bias_t = (const float*)d_in[4];   // [B,H,S,hd]
  float* out = (float*)d_out;                    // flat [B,H,S,hd] == [B,S,E]

  char* ws = (char*)d_ws;
  size_t off = 0;
  auto take = [&](size_t bytes) {
    size_t o = off;
    off = (off + bytes + 255) & ~(size_t)255;
    return o;
  };
  const size_t nX  = (size_t)Bn * Sn * En;
  const size_t nW  = (size_t)F3 * En;
  const size_t nQK = (size_t)BH * Sn * HD;
  const size_t nSS = (size_t)BH * Sn * Sn;

  bf16_t* xb   = (bf16_t*)(ws + take(nX  * 2));
  bf16_t* wb   = (bf16_t*)(ws + take(nW  * 2));
  float*  qkvf = (float*) (ws + take((size_t)Bn * Sn * F3 * 4));
  bf16_t* qp   = (bf16_t*)(ws + take(nQK * 2));
  bf16_t* kp   = (bf16_t*)(ws + take(nQK * 2));
  bf16_t* vp   = (bf16_t*)(ws + take(nQK * 2));
  bf16_t* wopT = (bf16_t*)(ws + take(nQK * 2));
  bf16_t* qm   = (bf16_t*)(ws + take(nSS * 2));
  bf16_t* km   = (bf16_t*)(ws + take(nSS * 2));
  bf16_t* vtm  = (bf16_t*)(ws + take(nSS * 2));
  bf16_t* svm  = (bf16_t*)(ws + take(nSS * 2));
  bf16_t* zm   = (bf16_t*)(ws + take(nSS * 2));

  // 1) bf16 conversions
  f32_to_bf16<<<dim3((unsigned)((nX + 255) / 256)), 256, 0, stream>>>(x, xb, (int)nX);
  f32_to_bf16<<<dim3((unsigned)((nW + 255) / 256)), 256, 0, stream>>>(Wqkv, wb, (int)nW);

  // 2) qkv projection: [2048,1536] = x[2048,512] @ Wqkv[1536,512]^T
  wmma_gemm<float, false><<<dim3(96, 1), 256, 0, stream>>>(
      xb, wb, qkvf, Bn * Sn, F3, En, 0L, 0L, 0L, 1.0f);

  // 3) pruning
  prune_qkv<<<dim3(BH), 256, 0, stream>>>(qkvf, bqkv, bias_t, qp, kp, vp);
  prune_wo<<<dim3(BH), 256, 0, stream>>>(WO, bias_t, wopT);

  const long sQK = (long)Sn * HD, sSS = (long)Sn * Sn;
  // 4) q_ = (x_ @ qp^T)/8 ; k_ = x_ @ kp^T ; v_^T = vp @ x_^T   (K = 64)
  wmma_gemm<bf16_t, false><<<dim3(8, BH), 256, 0, stream>>>(
      xb, qp, qm, Sn, Sn, HD, sQK, sQK, sSS, 0.125f);
  wmma_gemm<bf16_t, false><<<dim3(8, BH), 256, 0, stream>>>(
      xb, kp, km, Sn, Sn, HD, sQK, sQK, sSS, 1.0f);
  wmma_gemm<bf16_t, false><<<dim3(8, BH), 256, 0, stream>>>(
      vp, xb, vtm, Sn, Sn, HD, sQK, sQK, sSS, 1.0f);

  // 5) SxSxS matmuls: LDS-staged async double-buffered WMMA
  //    scale_values = q_ @ k_^T ; z = sv @ v_ (NaN->0)
  wmma_gemm_lds<false><<<dim3(8, BH), 256, 0, stream>>>(qm, km, svm, 1.0f);
  wmma_gemm_lds<true ><<<dim3(8, BH), 256, 0, stream>>>(svm, vtm, zm, 1.0f);

  // 6) output = z @ WO_p -> f32 straight into d_out (flat [B,H,S,hd])
  wmma_gemm<float, false><<<dim3(1, BH), 256, 0, stream>>>(
      zm, wopT, out, Sn, HD, Sn, sSS, (long)HD * Sn, sQK, 1.0f);

  (void)n_in; (void)in_sizes; (void)out_size; (void)ws_size;
}
